// EGNNLayer_18537078849835
// MI455X (gfx1250) — compile-verified, matrix operand order checked by reading, not run on hardware
//
#include <hip/hip_runtime.h>
#include <hip/hip_bf16.h>

#define H       128
#define KE      288   // edge-MLP input K: 280 padded to 9*32
#define KN      256   // node-MLP input K
#define NRBF    20
#define EFEAT   4
#define NNODES  50000
#define NEDGES  800000
#define ET      32    // edges per block (2 M-tiles of 16)

typedef __attribute__((ext_vector_type(16))) __bf16 v16bf;
typedef __attribute__((ext_vector_type(8)))  float  v8f;

union FragU { v16bf v; uint4 q[2]; };

__device__ __forceinline__ float siluf(float v)  { return v / (1.f + __expf(-v)); }
__device__ __forceinline__ float sigmf_(float v) { return 1.f / (1.f + __expf(-v)); }

// ---- A fragment: 16x32 bf16 tile from row-major LDS (stride in elements) ----
// Lane: M = lane&15 (+tile*16), h = lane>>4. elems 0..7 = A[M][k0+8h..+7],
//                                            elems 8..15 = A[M][k0+16+8h..+7].
__device__ __forceinline__ v16bf load_afrag(const __bf16* A, int stride,
                                            int k0, int lane, int mtile) {
  int m = (lane & 15) + mtile * 16, hh = lane >> 4;
  FragU f;
  f.q[0] = *(const uint4*)(A + m * stride + k0 + 8 * hh);
  f.q[1] = *(const uint4*)(A + m * stride + k0 + 16 + 8 * hh);
  return f.v;
}

// ---- B fragment: 32x16 bf16, pre-packed so each lane reads 32 contiguous B ----
__device__ __forceinline__ v16bf load_bfrag(const __bf16* __restrict__ Wp,
                                            int kstep, int npanel, int lane) {
  FragU f;
  const uint4* p = (const uint4*)(Wp + ((((kstep * 8 + npanel) * 32) + lane) << 4));
  f.q[0] = p[0]; f.q[1] = p[1];
  return f.v;
}

// Pack W[Korig][128] (row-major f32) -> bf16 WMMA-B fragment layout, K padded to KP.
__global__ void pack_weight(const float* __restrict__ W, __bf16* __restrict__ dst,
                            int Korig, int KP) {
  int id = blockIdx.x * blockDim.x + threadIdx.x;
  if (id >= KP * H) return;
  int elem   = id & 15;
  int lane   = (id >> 4) & 31;
  int npanel = (id >> 9) & 7;
  int kstep  = id >> 12;
  int hh = lane >> 4;
  int n  = npanel * 16 + (lane & 15);
  int koff = (elem < 8) ? (8 * hh + elem) : (16 + 8 * hh + (elem - 8));
  int k = kstep * 32 + koff;
  dst[id] = (__bf16)((k < Korig) ? W[k * H + n] : 0.f);
}

__global__ void zero_out(float* p, int n) {
  int i = blockIdx.x * blockDim.x + threadIdx.x;
  if (i < n) p[i] = 0.f;
}

// ================= Edge kernel: 32 edges per block, 8 waves ==================
__global__ __launch_bounds__(256) void edge_kernel(
    const float* __restrict__ hfeat, const float* __restrict__ x,
    const float* __restrict__ eattr, const int* __restrict__ eidx,
    const __bf16* __restrict__ We1p, const float* __restrict__ be1,
    const __bf16* __restrict__ We2p, const float* __restrict__ be2,
    const float* __restrict__ Wg,  const float* __restrict__ bg,
    const __bf16* __restrict__ Wx1p, const float* __restrict__ bx1,
    const float* __restrict__ Wx2,
    float* __restrict__ agg, float* __restrict__ dx)
{
  __shared__ __align__(16) __bf16 A0[ET][KE];   // edge-MLP input
  __shared__ __align__(16) __bf16 A1[ET][H];    // layer-1 act; later: x-path act T
  __shared__ __align__(16) __bf16 A2[ET][H];    // m (bf16, x-path A)
  __shared__ float Mf[ET][H];                   // m (f32: gate dot + scatter)
  __shared__ float pg[ET][8], px[ET][8];
  __shared__ int   sSrc[ET], sDst[ET];
  __shared__ float sRel[ET][3], sInvR[ET], sR[ET], sG[ET], sC[ET];

  const int tid = threadIdx.x;
  const int lane = tid & 31, wave = tid >> 5;
  const int e0 = blockIdx.x * ET;

  // --- stage 1: per-edge geometry ---
  if (tid < ET) {
    int e = e0 + tid;
    int s = eidx[e], d = eidx[NEDGES + e];
    sSrc[tid] = s; sDst[tid] = d;
    float rx = x[d*3+0] - x[s*3+0];
    float ry = x[d*3+1] - x[s*3+1];
    float rz = x[d*3+2] - x[s*3+2];
    float r  = sqrtf(rx*rx + ry*ry + rz*rz + 1e-8f);
    sRel[tid][0] = rx; sRel[tid][1] = ry; sRel[tid][2] = rz;
    sR[tid] = r; sInvR[tid] = 1.f / (r + 1.f);
  }
  __syncthreads();

  // --- stage 2: build A0 = [h[dst] | h[src] | rbf | edge_attr | 0pad] ---
  {
    int i = tid >> 3, kk = tid & 7;           // 8 threads per edge row
    int e = e0 + i, d = sDst[i], s = sSrc[i];
    float r = sR[i];
    const float spacing = 10.0f / 19.0f;
    const float coeff = -0.5f / (spacing * spacing);
    #pragma unroll
    for (int c = 0; c < KE / 8; ++c) {
      int k = kk + 8 * c;
      float v;
      if (k < H)            v = hfeat[d * H + k];
      else if (k < 2 * H)   v = hfeat[s * H + (k - H)];
      else if (k < 2*H + NRBF) {
        float t = r - spacing * (float)(k - 2 * H);
        v = __expf(coeff * t * t);
      } else if (k < 2*H + NRBF + EFEAT) {
        v = eattr[e * EFEAT + (k - (2 * H + NRBF))];
      } else v = 0.f;
      A0[i][k] = (__bf16)v;
    }
  }
  __syncthreads();

  const int n  = wave * 16 + (lane & 15);   // this lane's output column
  const int mb = 8 * (lane >> 4);           // row base within an M-tile

  // --- GEMM 1: [32x288] @ We1 -> silu -> A1 (B loaded once, 2 WMMAs per k) ---
  v8f c0 = {}, c1 = {};
  #pragma unroll
  for (int ks = 0; ks < KE / 32; ++ks) {
    v16bf b  = load_bfrag(We1p, ks, wave, lane);
    v16bf a0 = load_afrag(&A0[0][0], KE, ks * 32, lane, 0);
    v16bf a1 = load_afrag(&A0[0][0], KE, ks * 32, lane, 1);
    c0 = __builtin_amdgcn_wmma_f32_16x16x32_bf16(false, a0, false, b, (short)0, c0, false, false);
    c1 = __builtin_amdgcn_wmma_f32_16x16x32_bf16(false, a1, false, b, (short)0, c1, false, false);
  }
  {
    float bias = be1[n];
    #pragma unroll
    for (int rr = 0; rr < 8; ++rr) {
      A1[mb + rr][n]      = (__bf16)siluf(c0[rr] + bias);
      A1[16 + mb + rr][n] = (__bf16)siluf(c1[rr] + bias);
    }
  }
  __syncthreads();

  // --- GEMM 2: @ We2 -> silu -> m (Mf f32 + A2 bf16) ---
  c0 = (v8f){}; c1 = (v8f){};
  #pragma unroll
  for (int ks = 0; ks < H / 32; ++ks) {
    v16bf b  = load_bfrag(We2p, ks, wave, lane);
    v16bf a0 = load_afrag(&A1[0][0], H, ks * 32, lane, 0);
    v16bf a1 = load_afrag(&A1[0][0], H, ks * 32, lane, 1);
    c0 = __builtin_amdgcn_wmma_f32_16x16x32_bf16(false, a0, false, b, (short)0, c0, false, false);
    c1 = __builtin_amdgcn_wmma_f32_16x16x32_bf16(false, a1, false, b, (short)0, c1, false, false);
  }
  {
    float bias = be2[n];
    #pragma unroll
    for (int rr = 0; rr < 8; ++rr) {
      float v0 = siluf(c0[rr] + bias);
      float v1 = siluf(c1[rr] + bias);
      Mf[mb + rr][n] = v0;      A2[mb + rr][n]      = (__bf16)v0;
      Mf[16 + mb + rr][n] = v1; A2[16 + mb + rr][n] = (__bf16)v1;
    }
  }
  __syncthreads();   // also guarantees all GEMM-2 reads of A1 are done

  // --- GEMM 3 (x-path): silu(m @ Wx1 + bx1) -> T, stored bf16 into A1 ---
  c0 = (v8f){}; c1 = (v8f){};
  #pragma unroll
  for (int ks = 0; ks < H / 32; ++ks) {
    v16bf b  = load_bfrag(Wx1p, ks, wave, lane);
    v16bf a0 = load_afrag(&A2[0][0], H, ks * 32, lane, 0);
    v16bf a1 = load_afrag(&A2[0][0], H, ks * 32, lane, 1);
    c0 = __builtin_amdgcn_wmma_f32_16x16x32_bf16(false, a0, false, b, (short)0, c0, false, false);
    c1 = __builtin_amdgcn_wmma_f32_16x16x32_bf16(false, a1, false, b, (short)0, c1, false, false);
  }
  {
    float bias = bx1[n];
    #pragma unroll
    for (int rr = 0; rr < 8; ++rr) {
      A1[mb + rr][n]      = (__bf16)siluf(c0[rr] + bias);
      A1[16 + mb + rr][n] = (__bf16)siluf(c1[rr] + bias);
    }
  }
  __syncthreads();

  // --- gate + coef dots (128-dot per edge; VALU) ---
  {
    int i = tid >> 3, j0 = tid & 7;
    float ag = 0.f, ax = 0.f;
    #pragma unroll
    for (int c = 0; c < 16; ++c) {
      int j = j0 + 8 * c;
      ag += Mf[i][j] * Wg[j];
      ax += (float)A1[i][j] * Wx2[j];
    }
    pg[i][j0] = ag; px[i][j0] = ax;
  }
  __syncthreads();
  if (tid < ET) {
    float sg = 0.f, sx = 0.f;
    #pragma unroll
    for (int j = 0; j < 8; ++j) { sg += pg[tid][j]; sx += px[tid][j]; }
    sG[tid] = sigmf_(sg + bg[0]);
    sC[tid] = tanhf(sx);
  }
  __syncthreads();

  // --- scatter: segment_sum via f32 atomics (L2-resident accumulators) ---
  {
    int i = tid >> 3, j0 = tid & 7;
    int d = sDst[i];
    float g = sG[i];
    #pragma unroll
    for (int c = 0; c < 16; ++c) {
      int j = j0 + 8 * c;
      atomicAdd(&agg[d * H + j], Mf[i][j] * g);
    }
  }
  if (tid < 3 * ET) {
    int i = tid / 3, comp = tid % 3;
    atomicAdd(&dx[sDst[i] * 3 + comp], sRel[i][comp] * sInvR[i] * sC[i]);
  }
}

// ================= Node kernel: 16 nodes per block, 8 waves ==================
__global__ __launch_bounds__(256) void node_kernel(
    const float* __restrict__ hfeat, const float* __restrict__ x,
    const int* __restrict__ mask,
    const __bf16* __restrict__ Wn1p, const float* __restrict__ bn1,
    const __bf16* __restrict__ Wn2p, const float* __restrict__ bn2,
    float* __restrict__ hout /* in: agg, out: h_out */,
    float* __restrict__ xout /* in: dx,  out: x_out */)
{
  __shared__ __align__(16) __bf16 A0[16][KN];
  __shared__ __align__(16) __bf16 A1[16][H];
  __shared__ float sDx[16][3];

  const int tid = threadIdx.x;
  const int lane = tid & 31, wave = tid >> 5;
  const int n0 = blockIdx.x * 16;

  // --- build [agg | h] tile (read agg fully before overwriting h_out) ---
  {
    int i = tid >> 4, kk = tid & 15;
    int node = n0 + i;
    #pragma unroll
    for (int c = 0; c < KN / 16; ++c) {
      int k = kk + 16 * c;
      float v = (k < H) ? hout[node * H + k] : hfeat[node * H + (k - H)];
      A0[i][k] = (__bf16)v;
    }
  }
  if (tid < 48) {
    int i = tid / 3, comp = tid % 3;
    sDx[i][comp] = xout[(n0 + i) * 3 + comp];
  }
  __syncthreads();

  const int n  = wave * 16 + (lane & 15);
  const int mb = 8 * (lane >> 4);

  v8f c1 = {};
  #pragma unroll
  for (int ks = 0; ks < KN / 32; ++ks) {
    v16bf a = load_afrag(&A0[0][0], KN, ks * 32, lane, 0);
    v16bf b = load_bfrag(Wn1p, ks, wave, lane);
    c1 = __builtin_amdgcn_wmma_f32_16x16x32_bf16(false, a, false, b, (short)0, c1, false, false);
  }
  {
    float bias = bn1[n];
    #pragma unroll
    for (int rr = 0; rr < 8; ++rr)
      A1[mb + rr][n] = (__bf16)siluf(c1[rr] + bias);
  }
  __syncthreads();

  v8f c2 = {};
  #pragma unroll
  for (int ks = 0; ks < H / 32; ++ks) {
    v16bf a = load_afrag(&A1[0][0], H, ks * 32, lane, 0);
    v16bf b = load_bfrag(Wn2p, ks, wave, lane);
    c2 = __builtin_amdgcn_wmma_f32_16x16x32_bf16(false, a, false, b, (short)0, c2, false, false);
  }
  {
    float bias = bn2[n];
    #pragma unroll
    for (int rr = 0; rr < 8; ++rr) {
      int node = n0 + mb + rr;
      hout[node * H + n] = hfeat[node * H + n] + c2[rr] + bias;
    }
  }
  if (tid < 48) {
    int i = tid / 3, comp = tid % 3;
    int node = n0 + i;
    xout[node * 3 + comp] = x[node * 3 + comp] + sDx[i][comp] * (float)mask[node];
  }
}

extern "C" void kernel_launch(void* const* d_in, const int* in_sizes, int n_in,
                              void* d_out, int out_size, void* d_ws, size_t ws_size,
                              hipStream_t stream) {
  (void)in_sizes; (void)n_in; (void)out_size; (void)ws_size;
  const float* hfeat = (const float*)d_in[0];
  const float* x     = (const float*)d_in[1];
  const float* eattr = (const float*)d_in[2];
  const int*   eidx  = (const int*)d_in[3];
  const int*   mask  = (const int*)d_in[4];
  const float* We1 = (const float*)d_in[5];
  const float* be1 = (const float*)d_in[6];
  const float* We2 = (const float*)d_in[7];
  const float* be2 = (const float*)d_in[8];
  const float* Wg  = (const float*)d_in[9];
  const float* bg  = (const float*)d_in[10];
  const float* Wn1 = (const float*)d_in[11];
  const float* bn1 = (const float*)d_in[12];
  const float* Wn2 = (const float*)d_in[13];
  const float* bn2 = (const float*)d_in[14];
  const float* Wx1 = (const float*)d_in[15];
  const float* bx1 = (const float*)d_in[16];
  const float* Wx2 = (const float*)d_in[17];

  __bf16* ws   = (__bf16*)d_ws;
  __bf16* We1p = ws;
  __bf16* We2p = We1p + KE * H;
  __bf16* Wx1p = We2p + H * H;
  __bf16* Wn1p = Wx1p + H * H;
  __bf16* Wn2p = Wn1p + KN * H;

  float* hout = (float*)d_out;          // agg accumulator, then h_out
  float* xout = hout + NNODES * H;      // dx accumulator, then x_out

  int ztot = NNODES * H + NNODES * 3;
  zero_out<<<(ztot + 255) / 256, 256, 0, stream>>>((float*)d_out, ztot);

  pack_weight<<<(KE * H + 255) / 256, 256, 0, stream>>>(We1, We1p, 2 * H + NRBF + EFEAT, KE);
  pack_weight<<<(H * H + 255) / 256, 256, 0, stream>>>(We2, We2p, H, H);
  pack_weight<<<(H * H + 255) / 256, 256, 0, stream>>>(Wx1, Wx1p, H, H);
  pack_weight<<<(KN * H + 255) / 256, 256, 0, stream>>>(Wn1, Wn1p, KN, KN);
  pack_weight<<<(H * H + 255) / 256, 256, 0, stream>>>(Wn2, Wn2p, H, H);

  edge_kernel<<<NEDGES / ET, 256, 0, stream>>>(hfeat, x, eattr, eidx,
      We1p, be1, We2p, be2, Wg, bg, Wx1p, bx1, Wx2, hout, xout);

  node_kernel<<<NNODES / 16, 256, 0, stream>>>(hfeat, x, mask,
      Wn1p, bn1, Wn2p, bn2, hout, xout);
}
